// Block_2233382994623
// MI455X (gfx1250) — matrix operand order repaired and checked by reading, not compiled, not run on hardware
//
#include <hip/hip_runtime.h>
#include <stdint.h>

// ---------------------------------------------------------------------------
// CDNA5 / gfx1250 equivariant 3D conv block.
//  stage 0: svt = concat(s, v, v⊗v) split to bf16 hi/lo planes,
//           layout [n][slot][hi[224] | lo[224]] (+1 zero slot per batch).
//  stage 1: conv kernel assembled, split hi/lo, stored as kb-contiguous
//           pre-swizzled 16x32 WMMA A-fragments (8KB per K-step).
//  stage 2: implicit-GEMM conv, tap-major K. 8-wave WGs share A via a
//           2x16KB double-buffered LDS pipeline (2 K-steps per stage);
//           B software-pipelined one stage ahead (plain b128 loads, no
//           unpack). v_wmma_f32_16x16x32_bf16, 3-term split compensation.
//  stage 3/4: variance reduction; normalize/bias/relu in place.
// ---------------------------------------------------------------------------

typedef __bf16 bf16_t;
typedef __attribute__((ext_vector_type(16))) __bf16 v16bf;
typedef __attribute__((ext_vector_type(8)))  float  v8f;

#define CI_PAD    224                     // 208 channels padded to 7*32
#define NTAP      343                     // 7^3 taps
#define NKB       2402                    // 343*7 K-blocks + 1 pad block
#define NPAIR     1201                    // K-steps processed 2 per stage
#define NSLOT     32768                   // 32^3 input voxels
#define SLOT_B    (CI_PAD * 4)            // 896 bytes per slot (hi448|lo448)
#define BATCH_SLOTS (NSLOT + 1)           // +1 zero slot (OOB target)
#define SAFE_OFF  (NSLOT * SLOT_B)        // byte offset of zero slot
#define SVT_U16_PER_N ((size_t)BATCH_SLOTS * CI_PAD * 2)
#define SVT_U16_TOTAL (4 * SVT_U16_PER_N)
#define AFRAG_U16_PER_KB 4096             // 8KB per K-step
#define AFRAG_U16_TOTAL ((size_t)NKB * AFRAG_U16_PER_KB)
#define OUT_ELEMS 1048576

#define WMMA_BF16(A, B, C) \
    __builtin_amdgcn_wmma_f32_16x16x32_bf16(false, (A), false, (B), (short)0, (C), false, false)

union BFrag { uint4 q[2]; v16bf v; };

// -------------------------------------------------------------------
// Stage 0: svt hi/lo planes per spatial slot.
// -------------------------------------------------------------------
__global__ void build_svt(const float* __restrict__ sv, uint16_t* __restrict__ svtP)
{
    size_t idx = (size_t)blockIdx.x * 256 + threadIdx.x;   // (n, s, cip)
    if (idx >= (size_t)4 * BATCH_SLOTS * CI_PAD) return;
    int cip = (int)(idx % CI_PAD);
    size_t rest = idx / CI_PAD;
    int s = (int)(rest % BATCH_SLOTS);
    int n = (int)(rest / BATCH_SLOTS);

    float val = 0.f;
    if (s < NSLOT && cip < 208) {
        const float* __restrict__ svn = sv + (size_t)n * 64 * NSLOT;
        if (cip < 64) {
            val = svn[(size_t)cip * NSLOT + s];
        } else {                                    // tensor channel v_i * v_j
            int c = cip - 64;
            int u = c / 9, r2 = c - u * 9;
            int i = r2 / 3, j = r2 - i * 3;
            val = svn[(size_t)(16 + u*3 + i) * NSLOT + s] *
                  svn[(size_t)(16 + u*3 + j) * NSLOT + s];
        }
    }
    bf16_t h = (bf16_t)val;
    bf16_t l = (bf16_t)(val - (float)h);
    uint16_t* slot = svtP + ((size_t)n * BATCH_SLOTS + s) * (CI_PAD * 2);
    slot[cip]          = __builtin_bit_cast(uint16_t, h);
    slot[CI_PAD + cip] = __builtin_bit_cast(uint16_t, l);
}

// -------------------------------------------------------------------
// Stage 1: kernel assembly -> kb-contiguous A-fragments (hi then lo).
// element index in a kb block: (mt*2+h)*512 + lane*16 + e
// covers t in [0,344): t==343 writes the zero pad block (kb 2401).
// -------------------------------------------------------------------
__global__ void build_kfrag(
    const float* __restrict__ b_ss, const float* __restrict__ w_ss,
    const float* __restrict__ b_sv, const float* __restrict__ w_sv,
    const float* __restrict__ b_st, const float* __restrict__ w_st,
    const float* __restrict__ b_vs, const float* __restrict__ w_vs,
    const float* __restrict__ b_vv, const float* __restrict__ w_vv,
    const float* __restrict__ b_vt, const float* __restrict__ w_vt,
    uint16_t* __restrict__ Afrag)
{
    int idx = blockIdx.x * 256 + threadIdx.x;     // 64*344*224 exact
    int cip = idx % CI_PAD;
    int t   = (idx / CI_PAD) % 344;
    int co  = idx / (CI_PAD * 344);

    float val = 0.f;
    if (cip < 208 && t < NTAP) {
        int ci = cip;
        if (co < 16) {
            int u = co;
            if (ci < 16) {
                for (int b = 0; b < 3; ++b)
                    val += w_ss[(u*16 + ci)*3 + b] * b_ss[b*343 + t];
            } else if (ci < 64) {
                int v = (ci-16)/3, j = (ci-16)%3;
                for (int b = 0; b < 3; ++b)
                    val += w_sv[(u*16 + v)*3 + b] * b_sv[(b*3 + j)*343 + t];
            } else {
                int v = (ci-64)/9, j = (ci-64)%9;
                for (int b = 0; b < 9; ++b)
                    val += w_st[(u*16 + v)*9 + b] * b_st[(b*9 + j)*343 + t];
            }
        } else {
            int u = (co-16)/3, i = (co-16)%3;
            if (ci < 16) {
                for (int b = 0; b < 3; ++b)
                    val += w_vs[(u*16 + ci)*3 + b] * b_vs[(b*3 + i)*343 + t];
            } else if (ci < 64) {
                int v = (ci-16)/3, j = (ci-16)%3;
                for (int b = 0; b < 9; ++b)
                    val += w_vv[(u*16 + v)*9 + b] * b_vv[((b*3 + i)*3 + j)*343 + t];
            } else {
                int v = (ci-64)/9, j = (ci-64)%9;
                for (int b = 0; b < 21; ++b)
                    val += w_vt[(u*16 + v)*21 + b] * b_vt[((b*3 + i)*9 + j)*343 + t];
            }
        }
    }
    bf16_t h = (bf16_t)val;
    bf16_t l = (bf16_t)(val - (float)h);

    int kb     = t * 7 + (cip >> 5);
    int klocal = cip & 31;
    int lh = klocal >> 4;
    int e  = klocal & 15;
    int mt = co >> 4, m = co & 15;
    int ln = m + 16*lh;
    size_t base = (size_t)kb * AFRAG_U16_PER_KB;
    Afrag[base + (size_t)(mt*2 + 0)*512 + ln*16 + e] = __builtin_bit_cast(uint16_t, h);
    Afrag[base + (size_t)(mt*2 + 1)*512 + ln*16 + e] = __builtin_bit_cast(uint16_t, l);
}

// -------------------------------------------------------------------
// Stage 2: conv. 256-thread WG = 8 waves, each wave: M=64 x N=16 tile.
// -------------------------------------------------------------------
__global__ __launch_bounds__(256)
void conv_wmma(const uint16_t* __restrict__ svtP,
               const uint16_t* __restrict__ Afrag,
               float* __restrict__ y)
{
    __shared__ __align__(32) uint4 Albuf[2][1024];  // 2 x 16KB (2 K-steps each)

    const int tid  = threadIdx.x;
    const int lane = tid & 31;
    const int wv   = tid >> 5;
    const int p    = blockIdx.x * 128 + wv * 16 + (lane & 15);
    const int n    = p >> 12;                        // uniform per WG
    const int sp   = p & 4095;
    const int od = sp >> 8, oh = (sp >> 4) & 15, ow = sp & 15;
    const int iz0 = od*2 - 3, iy0 = oh*2 - 3, ix0 = ow*2 - 3;

    uint32_t vzm = 0, vym = 0, vxm = 0;
#pragma unroll
    for (int k = 0; k < 7; ++k) {
        if ((unsigned)(iz0 + k) < 32u) vzm |= 1u << k;
        if ((unsigned)(iy0 + k) < 32u) vym |= 1u << k;
        if ((unsigned)(ix0 + k) < 32u) vxm |= 1u << k;
    }
    const int sbase896 = (iz0*1024 + iy0*32 + ix0) * 896;
    const int khalf2   = (lane >> 4) * 32;           // bf16 K-half byte offset
    const uint8_t* __restrict__ svtN = (const uint8_t*)svtP + (size_t)n * SVT_U16_PER_N * 2;
    const uint4*   __restrict__ Ag   = (const uint4*)Afrag;

    // B loader: tap-major K-block kb -> bhi/blo fragments (pure b128 loads)
    auto loadB = [&](int kb, BFrag& BH, BFrag& BL) {
        const int t  = kb / 7;
        const int cb = kb - t*7;
        const int kd = t / 49; const int rr = t - kd*49;
        const int kh = rr / 7; const int kw = rr - kh*7;
        const int uni = (kd*1024 + kh*32 + kw) * 896 + cb*64;
        const uint32_t vbit = (vzm >> kd) & (vym >> kh) & (vxm >> kw) & 1u;
        int voff = sbase896 + uni;
        voff = vbit ? voff : (int)SAFE_OFF;          // OOB -> zero slot
        voff += khalf2;
        const uint8_t* bp = svtN + voff;
        BH.q[0] = *(const uint4*)(bp);
        BH.q[1] = *(const uint4*)(bp + 16);
        BL.q[0] = *(const uint4*)(bp + 448);
        BL.q[1] = *(const uint4*)(bp + 464);
    };

    v8f acc0 = {}, acc1 = {}, acc2 = {}, acc3 = {};

    // prologue: stage pair 0 (A -> LDS buf0, B -> regs)
    BFrag b0h, b0l, b1h, b1l;
    loadB(0, b0h, b0l);
    loadB(1, b1h, b1l);
#pragma unroll
    for (int i = 0; i < 4; ++i) Albuf[0][tid + i*256] = Ag[tid + i*256];
    __syncthreads();

    for (int kp = 0; kp < NPAIR; ++kp) {
        const int cur = kp & 1;
        const bool nx = (kp + 1) < NPAIR;

        // ---- prefetch next stage: A (global->VGPR), B (global->regs) ----
        uint4 pa0, pa1, pa2, pa3;
        BFrag n0h, n0l, n1h, n1l;
        if (nx) {
            const uint4* src = Ag + (size_t)(kp + 1) * 1024;
            pa0 = src[tid];       pa1 = src[tid + 256];
            pa2 = src[tid + 512]; pa3 = src[tid + 768];
            loadB(2*kp + 2, n0h, n0l);
            loadB(2*kp + 3, n1h, n1l);
        }
        if (kp + 2 < NPAIR)      // stream hint for the stage after next
            __builtin_prefetch((const uint8_t*)(Ag + (size_t)(kp + 2) * 1024) + tid * 64, 0, 0);

        // ---- consume the two staged K-steps ----
        const uint8_t* Ab = (const uint8_t*)&Albuf[cur][0];
#define AFR(half, mt, h) (*(const v16bf*)(Ab + (half)*8192 + ((mt)*2 + (h))*1024 + lane*32))
        {
            const v16bf a0h = AFR(0,0,0), a1h = AFR(0,1,0), a2h = AFR(0,2,0), a3h = AFR(0,3,0);
            const v16bf a0l = AFR(0,0,1), a1l = AFR(0,1,1), a2l = AFR(0,2,1), a3l = AFR(0,3,1);
            acc0 = WMMA_BF16(a0h, b0h.v, acc0);
            acc1 = WMMA_BF16(a1h, b0h.v, acc1);
            acc2 = WMMA_BF16(a2h, b0h.v, acc2);
            acc3 = WMMA_BF16(a3h, b0h.v, acc3);
            acc0 = WMMA_BF16(a0h, b0l.v, acc0);
            acc1 = WMMA_BF16(a1h, b0l.v, acc1);
            acc2 = WMMA_BF16(a2h, b0l.v, acc2);
            acc3 = WMMA_BF16(a3h, b0l.v, acc3);
            acc0 = WMMA_BF16(a0l, b0h.v, acc0);
            acc1 = WMMA_BF16(a1l, b0h.v, acc1);
            acc2 = WMMA_BF16(a2l, b0h.v, acc2);
            acc3 = WMMA_BF16(a3l, b0h.v, acc3);
        }
        {
            const v16bf a0h = AFR(1,0,0), a1h = AFR(1,1,0), a2h = AFR(1,2,0), a3h = AFR(1,3,0);
            const v16bf a0l = AFR(1,0,1), a1l = AFR(1,1,1), a2l = AFR(1,2,1), a3l = AFR(1,3,1);
            acc0 = WMMA_BF16(a0h, b1h.v, acc0);
            acc1 = WMMA_BF16(a1h, b1h.v, acc1);
            acc2 = WMMA_BF16(a2h, b1h.v, acc2);
            acc3 = WMMA_BF16(a3h, b1h.v, acc3);
            acc0 = WMMA_BF16(a0h, b1l.v, acc0);
            acc1 = WMMA_BF16(a1h, b1l.v, acc1);
            acc2 = WMMA_BF16(a2h, b1l.v, acc2);
            acc3 = WMMA_BF16(a3h, b1l.v, acc3);
            acc0 = WMMA_BF16(a0l, b1h.v, acc0);
            acc1 = WMMA_BF16(a1l, b1h.v, acc1);
            acc2 = WMMA_BF16(a2l, b1h.v, acc2);
            acc3 = WMMA_BF16(a3l, b1h.v, acc3);
        }
#undef AFR

        // ---- publish next A buffer, rotate B ----
        if (nx) {
            Albuf[cur ^ 1][tid]       = pa0;
            Albuf[cur ^ 1][tid + 256] = pa1;
            Albuf[cur ^ 1][tid + 512] = pa2;
            Albuf[cur ^ 1][tid + 768] = pa3;
            b0h = n0h; b0l = n0l; b1h = n1h; b1l = n1l;
        }
        __syncthreads();
    }

    // ---- store raw y: VGPR r -> M = r + 8*(lane>=16), N = lane%16 ----
    const int mrow = (lane >> 4) * 8;
    const size_t ob = (size_t)n * 262144 + sp;
#pragma unroll
    for (int r = 0; r < 8; ++r) {
        y[ob + (size_t)(     mrow + r) * 4096] = acc0[r];
        y[ob + (size_t)(16 + mrow + r) * 4096] = acc1[r];
        y[ob + (size_t)(32 + mrow + r) * 4096] = acc2[r];
        y[ob + (size_t)(48 + mrow + r) * 4096] = acc3[r];
    }
}

// -------------------------------------------------------------------
// Stage 3: variances. blocks 0..15 scalar channels, 16..31 vector groups.
// -------------------------------------------------------------------
__global__ void reduce_var(const float* __restrict__ y, float* __restrict__ var)
{
    __shared__ float red[256];
    const int b = blockIdx.x, tid = threadIdx.x;
    float s = 0.f;
    if (b < 16) {
        for (int i = tid; i < 16384; i += 256) {
            const int n = i >> 12, sp = i & 4095;
            const float v = y[(size_t)n * 262144 + (size_t)b * 4096 + sp];
            s += v * v;
        }
    } else {
        const int u = b - 16;
        for (int i = tid; i < 49152; i += 256) {
            const int n = i / 12288, rr = i % 12288;
            const int comp = rr >> 12, sp = rr & 4095;
            const int c = 16 + u * 3 + comp;
            const float v = y[(size_t)n * 262144 + (size_t)c * 4096 + sp];
            s += v * v;
        }
    }
    red[tid] = s;
    __syncthreads();
    for (int w = 128; w > 0; w >>= 1) {
        if (tid < w) red[tid] += red[tid + w];
        __syncthreads();
    }
    if (tid == 0) var[b] = red[0] / (b < 16 ? 16384.f : 49152.f);
}

// -------------------------------------------------------------------
// Stage 4: normalize (+bias+relu for scalars) in place.
// -------------------------------------------------------------------
__global__ void normalize_k(float* __restrict__ y, const float* __restrict__ var,
                            const float* __restrict__ bn_g_s,
                            const float* __restrict__ bn_g_v,
                            const float* __restrict__ bias_s)
{
    const int idx = blockIdx.x * blockDim.x + threadIdx.x;
    if (idx >= OUT_ELEMS) return;
    const int co = (idx >> 12) & 63;
    float v = y[idx];
    if (co < 16) {
        const float s = bn_g_s[co] / sqrtf(var[co] + 1e-5f);
        v = fmaxf(v * s + bias_s[co], 0.f);
    } else {
        const int u = (co - 16) / 3;
        const float s = bn_g_v[u] / sqrtf(var[16 + u] + 1e-5f);
        v = v * s;
    }
    y[idx] = v;
}

// -------------------------------------------------------------------
extern "C" void kernel_launch(void* const* d_in, const int* in_sizes, int n_in,
                              void* d_out, int out_size, void* d_ws, size_t ws_size,
                              hipStream_t stream)
{
    (void)in_sizes; (void)n_in; (void)out_size; (void)ws_size;

    const float* sv     = (const float*)d_in[0];
    const float* b_ss   = (const float*)d_in[1];
    const float* w_ss   = (const float*)d_in[2];
    const float* b_sv   = (const float*)d_in[3];
    const float* w_sv   = (const float*)d_in[4];
    const float* b_st   = (const float*)d_in[5];
    const float* w_st   = (const float*)d_in[6];
    const float* b_vs   = (const float*)d_in[7];
    const float* w_vs   = (const float*)d_in[8];
    const float* b_vv   = (const float*)d_in[9];
    const float* w_vv   = (const float*)d_in[10];
    const float* b_vt   = (const float*)d_in[11];
    const float* w_vt   = (const float*)d_in[12];
    const float* bn_g_s = (const float*)d_in[13];
    const float* bn_g_v = (const float*)d_in[14];
    const float* bias_s = (const float*)d_in[15];

    uint16_t* svtP  = (uint16_t*)d_ws;                       // 117.4 MB
    uint16_t* Afrag = svtP + SVT_U16_TOTAL;                  // 19.7 MB
    float*    var   = (float*)(Afrag + AFRAG_U16_TOTAL);
    float*    y     = (float*)d_out;

    build_svt<<<(int)(((size_t)4 * BATCH_SLOTS * CI_PAD + 255) / 256), 256, 0, stream>>>(sv, svtP);
    build_kfrag<<<19264, 256, 0, stream>>>(b_ss, w_ss, b_sv, w_sv, b_st, w_st,
                                           b_vs, w_vs, b_vv, w_vv, b_vt, w_vt,
                                           Afrag);
    conv_wmma<<<128, 256, 0, stream>>>(svtP, Afrag, y);
    reduce_var<<<32, 256, 0, stream>>>(y, var);
    normalize_k<<<OUT_ELEMS / 256, 256, 0, stream>>>(y, var, bn_g_s, bn_g_v, bias_s);
}
